// PairProjector_55396488184348
// MI455X (gfx1250) — compile-verified, matrix operand order checked by reading, not run on hardware
//
#include <hip/hip_runtime.h>

// ---------------------------------------------------------------------------
// CDNA5 (gfx1250) implementation. wave32; WMMA f32_16x16x32_f16 with fp32
// hi/lo split (Markidis): Ah*Bh + Ah*Bl + Al*Bh at f32 accumulate gives
// ~2^-22 relative error on the fast f16 matrix pipe. Operands are pre-split
// into hi/lo f16 planes so the GEMM hot loops do no conversion VALU.
// ---------------------------------------------------------------------------

typedef _Float16 v16h  __attribute__((ext_vector_type(16)));
typedef _Float16 f16x8 __attribute__((ext_vector_type(8)));
typedef float    v8f   __attribute__((ext_vector_type(8)));

#define TM 64          // M rows per block (sorted-entity tile)
#define TK 32          // K per WMMA step
#define NROLES 3
#define LDA 40         // LDS row stride in f16 (80B: 16B aligned, bank-skewed)

// ---------------- control kernels (role sort) ----------------

__global__ void k_init(int* __restrict__ ctrl, int* __restrict__ row_map, int n) {
  int i = blockIdx.x * blockDim.x + threadIdx.x;
  if (i < n) row_map[i] = -1;
  if (i < 16) ctrl[i] = 0;   // [0..2]=counts [4..6]=cursors [8..10]=poff
}

__global__ void k_hist(const int* __restrict__ roles, int E, int* __restrict__ ctrl) {
  int i = blockIdx.x * blockDim.x + threadIdx.x;
  if (i < E) atomicAdd(&ctrl[roles[i]], 1);
}

__global__ void k_offsets(int* __restrict__ ctrl, int* __restrict__ tile_role, int maxTiles) {
  __shared__ int s_off[NROLES], s_cnt[NROLES];
  if (threadIdx.x == 0) {
    int off = 0;
    for (int r = 0; r < NROLES; ++r) {
      ctrl[8 + r] = off; s_off[r] = off; s_cnt[r] = ctrl[r];
      off += ctrl[r];
      off = ((off + TM - 1) / TM) * TM;          // tile-align next segment
    }
  }
  __syncthreads();
  for (int t = threadIdx.x; t < maxTiles; t += blockDim.x) {
    int role = -1, base = t * TM;
    for (int r = 0; r < NROLES; ++r)
      if (base >= s_off[r] && base < s_off[r] + s_cnt[r]) role = r;
    tile_role[t] = role;
  }
}

__global__ void k_scatter(const int* __restrict__ roles, int E,
                          int* __restrict__ ctrl, int* __restrict__ row_map) {
  int i = blockIdx.x * blockDim.x + threadIdx.x;
  if (i >= E) return;
  int r = roles[i];
  int pos = ctrl[8 + r] + atomicAdd(&ctrl[4 + r], 1);
  row_map[pos] = i;
}

// ---------------- weight pre-split: fp32 -> hi/lo f16 planes ----------------

__global__ void k_split(const float* __restrict__ base, const int* __restrict__ layer_idx_p,
                        size_t perLayer, _Float16* __restrict__ hi,
                        _Float16* __restrict__ lo, int n) {
  int i = blockIdx.x * blockDim.x + threadIdx.x;
  if (i >= n) return;
  float x = base[(size_t)layer_idx_p[0] * perLayer + i];
  _Float16 h = (_Float16)x;
  hi[i] = h;
  lo[i] = (_Float16)(x - (float)h);
}

// ---------------- GEMM building blocks ----------------

// Cooperative load of a 64x32 f32 tile (row-gathered) -> split hi/lo f16 in LDS.
__device__ inline void load_tile_f32g(const float* __restrict__ src, int row0, int k0, int H,
                                      const int* __restrict__ row_map,
                                      _Float16* __restrict__ Shi, _Float16* __restrict__ Slo) {
  int t = threadIdx.x;           // 256 threads
  int r = t >> 2;                // 0..63
  int c = (t & 3) * 8;           // 0,8,16,24
  int srow = row_map[row0 + r];
  float vals[8];
  if (srow >= 0) {
    const float* p = src + (size_t)srow * H + (size_t)(k0 + c);
#pragma unroll
    for (int j = 0; j < 8; ++j) vals[j] = p[j];
  } else {
#pragma unroll
    for (int j = 0; j < 8; ++j) vals[j] = 0.0f;
  }
#pragma unroll
  for (int j = 0; j < 8; ++j) {
    _Float16 h = (_Float16)vals[j];
    Shi[r * LDA + c + j] = h;
    Slo[r * LDA + c + j] = (_Float16)(vals[j] - (float)h);
  }
}

// Cooperative copy of pre-split hi/lo f16 tiles (nrows x 32) -> LDS, no VALU.
__device__ inline void load_tile_f16(const _Float16* __restrict__ srcHi,
                                     const _Float16* __restrict__ srcLo,
                                     int row0, int k0, int H,
                                     _Float16* __restrict__ Shi, _Float16* __restrict__ Slo,
                                     int nrows) {
  for (int idx = threadIdx.x; idx < nrows * 4; idx += 256) {
    int r = idx >> 2, c = (idx & 3) * 8;
    size_t g = (size_t)(row0 + r) * H + (size_t)(k0 + c);
    *(f16x8*)&Shi[r * LDA + c] = *(const f16x8*)&srcHi[g];
    *(f16x8*)&Slo[r * LDA + c] = *(const f16x8*)&srcLo[g];
  }
}

// Per-lane WMMA fragment per ISA layout:
// lane<16: K{0..7,16..23}; lane>=16: K{8..15,24..31}, row = lane&15 (+rowbase).
__device__ inline v16h load_frag(const _Float16* __restrict__ S, int rowbase) {
  int lane = threadIdx.x & 31;
  int row  = rowbase + (lane & 15);
  int ko   = (lane >> 4) * 8;
  f16x8 a = *(const f16x8*)&S[row * LDA + ko];
  f16x8 b = *(const f16x8*)&S[row * LDA + ko + 16];
  return __builtin_shufflevector(a, b, 0,1,2,3,4,5,6,7,8,9,10,11,12,13,14,15);
}

// fp32-accurate product via 3 f16 WMMAs.
__device__ inline v8f mma3(v16h ah, v16h al, v16h bh, v16h bl, v8f c) {
  c = __builtin_amdgcn_wmma_f32_16x16x32_f16(false, ah, false, bh, (short)0, c, false, false);
  c = __builtin_amdgcn_wmma_f32_16x16x32_f16(false, ah, false, bl, (short)0, c, false, false);
  c = __builtin_amdgcn_wmma_f32_16x16x32_f16(false, al, false, bh, (short)0, c, false, false);
  return c;
}

// ---------------- stage 1: h = (v@W^T + Wb) * (r@U^T + Ub), gathered ----------------
// Block tile 64M x 64N, 8 waves each 16M x 32N, dual accumulators.

__global__ __launch_bounds__(256)
void k_stage1(const float* __restrict__ v_repr, const float* __restrict__ r_repr,
              const _Float16* __restrict__ Whi, const _Float16* __restrict__ Wlo,
              const _Float16* __restrict__ Uhi, const _Float16* __restrict__ Ulo,
              const float* __restrict__ Wb, const float* __restrict__ Ub,
              const int* __restrict__ layer_idx_p,
              const int* __restrict__ tile_role, const int* __restrict__ row_map,
              _Float16* __restrict__ hhi, _Float16* __restrict__ hlo, int H, int R) {
  __shared__ _Float16 sAvh[TM * LDA], sAvl[TM * LDA];
  __shared__ _Float16 sArh[TM * LDA], sArl[TM * LDA];
  __shared__ _Float16 sBwh[64 * LDA], sBwl[64 * LDA];
  __shared__ _Float16 sBuh[64 * LDA], sBul[64 * LDA];

  int tile = blockIdx.x;
  int role = tile_role[tile];
  if (role < 0) return;
  int layer = layer_idx_p[0];
  size_t wb = (size_t)role * H * H;      // pre-split planes hold the active layer only

  int m0 = tile * TM, n0 = blockIdx.y * 64;
  int wave = threadIdx.x >> 5;
  int wm = wave & 3, wn = wave >> 2;

  v8f accW[2] = {}; v8f accU[2] = {};
  for (int k0 = 0; k0 < H; k0 += TK) {
    load_tile_f32g(v_repr, m0, k0, H, row_map, sAvh, sAvl);
    load_tile_f32g(r_repr, m0, k0, H, row_map, sArh, sArl);
    load_tile_f16(Whi + wb, Wlo + wb, n0, k0, H, sBwh, sBwl, 64);
    load_tile_f16(Uhi + wb, Ulo + wb, n0, k0, H, sBuh, sBul, 64);
    __syncthreads();
    v16h avh = load_frag(sAvh, wm * 16), avl = load_frag(sAvl, wm * 16);
    v16h arh = load_frag(sArh, wm * 16), arl = load_frag(sArl, wm * 16);
#pragma unroll
    for (int s = 0; s < 2; ++s) {
      int nb = wn * 32 + s * 16;
      v16h bwh = load_frag(sBwh, nb), bwl = load_frag(sBwl, nb);
      v16h buh = load_frag(sBuh, nb), bul = load_frag(sBul, nb);
      accW[s] = mma3(avh, avl, bwh, bwl, accW[s]);
      accU[s] = mma3(arh, arl, buh, bul, accU[s]);
    }
    __syncthreads();
  }

  int lane = threadIdx.x & 31;
  const float* Wbl = Wb + ((size_t)layer * R + role) * H;
  const float* Ubl = Ub + ((size_t)layer * R + role) * H;
#pragma unroll
  for (int s = 0; s < 2; ++s) {
    int col = n0 + wn * 32 + s * 16 + (lane & 15);
    float bw = Wbl[col], bu = Ubl[col];
#pragma unroll
    for (int g = 0; g < 8; ++g) {
      int mrow = m0 + wm * 16 + g + (lane >> 4) * 8;
      float h = (accW[s][g] + bw) * (accU[s][g] + bu);
      h = (row_map[mrow] >= 0) ? h : 0.0f;
      _Float16 hh = (_Float16)h;
      size_t o = (size_t)mrow * H + col;
      hhi[o] = hh;
      hlo[o] = (_Float16)(h - (float)hh);
    }
  }
}

// ---------------- stages 2/3: block tile 64M x 128N, 8 waves each 32M x 32N ---------
// EPI==0: dst = relu(A@P^T + b), re-split to hi/lo planes (sorted -> sorted)
// EPI==1: out[row_map[row]] = A@P^T + b   (sorted -> scatter, f32)

template<int EPI>
__global__ __launch_bounds__(256)
void k_gemm(const _Float16* __restrict__ Ahi, const _Float16* __restrict__ Alo,
            const _Float16* __restrict__ Bhi, const _Float16* __restrict__ Blo,
            const float* __restrict__ Pb, const int* __restrict__ layer_idx_p,
            const int* __restrict__ tile_role, const int* __restrict__ row_map,
            float* __restrict__ outF,
            _Float16* __restrict__ dsthi, _Float16* __restrict__ dstlo,
            int H, int R) {
  __shared__ _Float16 sAh[TM * LDA],  sAl[TM * LDA];
  __shared__ _Float16 sBh[128 * LDA], sBl[128 * LDA];

  int tile = blockIdx.x;
  int role = tile_role[tile];
  if (role < 0) return;
  int layer = layer_idx_p[0];
  size_t pb = (size_t)role * H * H;

  int m0 = tile * TM, n0 = blockIdx.y * 128;
  int wave = threadIdx.x >> 5;
  int wm = wave & 1, wn = wave >> 1;     // 2 (M) x 4 (N) wave grid

  v8f acc[2][2] = {};
  for (int k0 = 0; k0 < H; k0 += TK) {
    load_tile_f16(Ahi,      Alo,      m0, k0, H, sAh, sAl, 64);
    load_tile_f16(Bhi + pb, Blo + pb, n0, k0, H, sBh, sBl, 128);
    __syncthreads();
    v16h ah[2], al[2], bh[2], bl[2];
#pragma unroll
    for (int i = 0; i < 2; ++i) {
      ah[i] = load_frag(sAh, wm * 32 + i * 16);
      al[i] = load_frag(sAl, wm * 32 + i * 16);
      bh[i] = load_frag(sBh, wn * 32 + i * 16);
      bl[i] = load_frag(sBl, wn * 32 + i * 16);
    }
#pragma unroll
    for (int i = 0; i < 2; ++i)
#pragma unroll
      for (int j = 0; j < 2; ++j)
        acc[i][j] = mma3(ah[i], al[i], bh[j], bl[j], acc[i][j]);
    __syncthreads();
  }

  int lane = threadIdx.x & 31;
  const float* Pbl = Pb + ((size_t)layer * R + role) * H;
#pragma unroll
  for (int i = 0; i < 2; ++i) {
#pragma unroll
    for (int j = 0; j < 2; ++j) {
      int col = n0 + wn * 32 + j * 16 + (lane & 15);
      float b = Pbl[col];
#pragma unroll
      for (int g = 0; g < 8; ++g) {
        int mrow = m0 + wm * 32 + i * 16 + g + (lane >> 4) * 8;
        float val = acc[i][j][g] + b;
        if (EPI == 0) {
          val = fmaxf(val, 0.0f);
          _Float16 vh = (_Float16)val;
          size_t o = (size_t)mrow * H + col;
          dsthi[o] = vh;
          dstlo[o] = (_Float16)(val - (float)vh);
        } else {
          int e = row_map[mrow];
          if (e >= 0) outF[(size_t)e * H + col] = val;
        }
      }
    }
  }
}

// ---------------- host launch ----------------

static inline size_t align256(size_t x) { return (x + 255) & ~(size_t)255; }

extern "C" void kernel_launch(void* const* d_in, const int* in_sizes, int n_in,
                              void* d_out, int out_size, void* d_ws, size_t ws_size,
                              hipStream_t stream) {
  const float* v_repr = (const float*)d_in[0];
  const float* r_repr = (const float*)d_in[1];
  const int*   roles  = (const int*)d_in[2];
  const float* W   = (const float*)d_in[3];
  const float* Wb  = (const float*)d_in[4];
  const float* U   = (const float*)d_in[5];
  const float* Ub  = (const float*)d_in[6];
  const float* P1  = (const float*)d_in[7];
  const float* P1b = (const float*)d_in[8];
  const float* P2  = (const float*)d_in[9];
  const float* P2b = (const float*)d_in[10];
  const int*   layer_idx = (const int*)d_in[11];
  float* out = (float*)d_out;

  const int E = in_sizes[2];
  const int H = in_sizes[0] / E;                // 512
  const int R = NROLES;
  const int maxTiles = (E + TM - 1) / TM + R;   // padded role segments
  const int padRows  = maxTiles * TM;
  const int nW       = R * H * H;               // per-layer weight slice elements
  const size_t perLayer = (size_t)nW;

  char* ws = (char*)d_ws;
  size_t off = 0;
  int* ctrl      = (int*)(ws + off); off += 256;
  int* tile_role = (int*)(ws + off); off = align256(off + (size_t)maxTiles * 4);
  int* row_map   = (int*)(ws + off); off = align256(off + (size_t)padRows * 4);

  _Float16* wplane[8];                          // Whi Wlo Uhi Ulo P1hi P1lo P2hi P2lo
  for (int i = 0; i < 8; ++i) {
    wplane[i] = (_Float16*)(ws + off);
    off = align256(off + (size_t)nW * 2);
  }
  size_t actBytes = (size_t)padRows * H * 2;
  _Float16* hhi = (_Float16*)(ws + off); off = align256(off + actBytes);
  _Float16* hlo = (_Float16*)(ws + off); off = align256(off + actBytes);
  _Float16* thi = (_Float16*)(ws + off); off = align256(off + actBytes);
  _Float16* tlo = (_Float16*)(ws + off); off = align256(off + actBytes);

  int tpb = 256;
  k_init   <<<(padRows + tpb - 1) / tpb, tpb, 0, stream>>>(ctrl, row_map, padRows);
  k_hist   <<<(E + tpb - 1) / tpb,       tpb, 0, stream>>>(roles, E, ctrl);
  k_offsets<<<1,                          tpb, 0, stream>>>(ctrl, tile_role, maxTiles);
  k_scatter<<<(E + tpb - 1) / tpb,       tpb, 0, stream>>>(roles, E, ctrl, row_map);

  int sgrid = (nW + tpb - 1) / tpb;
  k_split<<<sgrid, tpb, 0, stream>>>(W,  layer_idx, perLayer, wplane[0], wplane[1], nW);
  k_split<<<sgrid, tpb, 0, stream>>>(U,  layer_idx, perLayer, wplane[2], wplane[3], nW);
  k_split<<<sgrid, tpb, 0, stream>>>(P1, layer_idx, perLayer, wplane[4], wplane[5], nW);
  k_split<<<sgrid, tpb, 0, stream>>>(P2, layer_idx, perLayer, wplane[6], wplane[7], nW);

  dim3 g1(maxTiles, H / 64);
  k_stage1<<<g1, tpb, 0, stream>>>(v_repr, r_repr, wplane[0], wplane[1], wplane[2], wplane[3],
                                   Wb, Ub, layer_idx, tile_role, row_map, hhi, hlo, H, R);
  dim3 g2(maxTiles, H / 128);
  k_gemm<0><<<g2, tpb, 0, stream>>>(hhi, hlo, wplane[4], wplane[5], P1b, layer_idx,
                                    tile_role, row_map, nullptr, thi, tlo, H, R);
  k_gemm<1><<<g2, tpb, 0, stream>>>(thi, tlo, wplane[6], wplane[7], P2b, layer_idx,
                                    tile_role, row_map, out, nullptr, nullptr, H, R);
}
